// SelfAttention_63342177681707
// MI455X (gfx1250) — matrix operand order repaired and checked by reading, not compiled, not run on hardware
//
#include <hip/hip_runtime.h>
#include <hip/hip_bf16.h>
#include <stdint.h>

// ---------------- problem constants ----------------
#define B_    4
#define S_    1024
#define D_    4096
#define HQ_   32
#define HKV_  8
#define HD_   128
// N_REP = HQ_/HKV_ = 4

typedef unsigned short u16;
typedef __attribute__((ext_vector_type(16))) __bf16 v16bf;
typedef __attribute__((ext_vector_type(8)))  float  v8f;

// ---------------- CDNA5 async global->LDS path (inline asm, GV mode) ----------------
// global_load_async_to_lds_b128 vdst(LDS byte offset), vaddr(64-bit global), off
// Tracked by ASYNCcnt; drained with s_wait_asynccnt before the workgroup barrier.
#ifndef USE_ASYNC_LDS
#define USE_ASYNC_LDS 1
#endif

typedef __attribute__((address_space(3))) u16 lds_u16;

__device__ __forceinline__ void copy16_g2l(const u16* __restrict__ g, u16* l) {
#if USE_ASYNC_LDS
    lds_u16* lp = (lds_u16*)l; // addrspace cast: value == LDS byte offset (32-bit)
    asm volatile("global_load_async_to_lds_b128 %0, %1, off"
                 :
                 : "v"(lp), "v"(g)
                 : "memory");
#else
    *(uint4*)l = *(const uint4*)g;
#endif
}

__device__ __forceinline__ void wait_async_copies() {
#if USE_ASYNC_LDS
    asm volatile("s_wait_asynccnt 0x0" ::: "memory");
#endif
}

// ---------------- bf16 helpers (bit-level, RNE) ----------------
__device__ __forceinline__ u16 f2bf(float f) {
    unsigned int u = __float_as_uint(f);
    unsigned int r = u + 0x7FFFu + ((u >> 16) & 1u);
    return (u16)(r >> 16);
}
__device__ __forceinline__ float bf2f(u16 h) {
    return __uint_as_float(((unsigned int)h) << 16);
}

// Build a 16-element bf16 fragment from two 16-byte chunks.
__device__ __forceinline__ v16bf ldfrag(const u16* c0, const u16* c1) {
    union { uint4 u[2]; v16bf v; } x;
    x.u[0] = *(const uint4*)c0;
    x.u[1] = *(const uint4*)c1;
    return x.v;
}

__device__ __forceinline__ v8f wmma_bf16(v16bf a, v16bf b, v8f c) {
    return __builtin_amdgcn_wmma_f32_16x16x32_bf16(false, a, false, b, (short)0, c, false, false);
}

// ---------------- f32 -> bf16 convert ----------------
__global__ void cvt_f32_bf16(const float* __restrict__ in, u16* __restrict__ out, int n) {
    int i = blockIdx.x * blockDim.x + threadIdx.x;
    if (i < n) out[i] = f2bf(in[i]);
}

// ---------------- in-place interleaved RoPE on bf16 [B*S, H, HD] ----------------
__global__ void rope_bf16(u16* __restrict__ x, const float* __restrict__ cosf,
                          const float* __restrict__ sinf, int H, int total) {
    int i = blockIdx.x * blockDim.x + threadIdx.x;
    if (i >= total) return;
    int d2 = i & 63;                 // HD/2 = 64
    int h  = (i >> 6) % H;
    int bs = i / (64 * H);
    int s  = bs & (S_ - 1);
    size_t base = ((size_t)bs * H + h) * HD_ + 2 * d2;
    float xr = bf2f(x[base]);
    float xi = bf2f(x[base + 1]);
    float c  = cosf[s * 64 + d2];
    float sn = sinf[s * 64 + d2];
    x[base]     = f2bf(xr * c - xi * sn);
    x[base + 1] = f2bf(xr * sn + xi * c);
}

// ---------------- WMMA GEMM: C[M,N] = A[M,K] * B[N,K]^T ----------------
// A, B bf16 row-major (K contiguous). Output f32 or bf16. M,N mult of 128; K mult of 64.
template <bool OUT_BF16>
__global__ __launch_bounds__(256) void gemm_absT(const u16* __restrict__ A,
                                                 const u16* __restrict__ Bw,
                                                 void* __restrict__ Cout,
                                                 int M, int N, int K) {
    __shared__ __align__(16) u16 sA[128 * 72]; // 128x64, +8 pad
    __shared__ __align__(16) u16 sB[128 * 72];
    const int t    = threadIdx.x;
    const int lane = t & 31;
    const int wave = t >> 5;
    const int ln   = lane & 15;
    const bool hi  = lane >= 16;
    const int m0 = blockIdx.y * 128;
    const int n0 = blockIdx.x * 128;
    const int wm = (wave >> 1) * 32; // 0,32,64,96
    const int wn = (wave & 1) * 64;  // 0,64

    v8f acc[2][4];
#pragma unroll
    for (int i = 0; i < 2; ++i)
#pragma unroll
        for (int j = 0; j < 4; ++j)
#pragma unroll
            for (int e = 0; e < 8; ++e) acc[i][j][e] = 0.f;

    for (int kt = 0; kt < K; kt += 64) {
        // cooperative tile staging: 128x64 bf16 each, 16B per thread per pass
#pragma unroll
        for (int p = 0; p < 4; ++p) {
            int e = (p * 256 + t) * 8;
            int r = e >> 6, c = e & 63;
            copy16_g2l(&A[(size_t)(m0 + r) * K + kt + c], &sA[r * 72 + c]);
            copy16_g2l(&Bw[(size_t)(n0 + r) * K + kt + c], &sB[r * 72 + c]);
        }
        wait_async_copies();
        __syncthreads();

#pragma unroll
        for (int kk = 0; kk < 64; kk += 32) {
            // A fragment: lane<16 holds K{0..7,16..23}, lane>=16 holds K{8..15,24..31}
            v16bf af[2];
#pragma unroll
            for (int i = 0; i < 2; ++i) {
                const u16* base = &sA[(wm + i * 16 + ln) * 72 + kk];
                int k0 = hi ? 8 : 0;
                af[i] = ldfrag(base + k0, base + k0 + 16);
            }
            // B fragment: lane n = column n; lane<16 K{0..7,16..23}, lane>=16 K{8..15,24..31}
            v16bf bfg[4];
#pragma unroll
            for (int j = 0; j < 4; ++j) {
                const u16* base = &sB[(wn + j * 16 + ln) * 72 + kk];
                int k0 = hi ? 8 : 0;
                bfg[j] = ldfrag(base + k0, base + k0 + 16);
            }
#pragma unroll
            for (int i = 0; i < 2; ++i)
#pragma unroll
                for (int j = 0; j < 4; ++j)
                    acc[i][j] = wmma_bf16(af[i], bfg[j], acc[i][j]);
        }
        __syncthreads();
    }

    // store: D layout — vgpr e: rows e (lanes 0-15) / e+8 (lanes 16-31), col = ln
#pragma unroll
    for (int i = 0; i < 2; ++i)
#pragma unroll
        for (int j = 0; j < 4; ++j)
#pragma unroll
            for (int e = 0; e < 8; ++e) {
                int row = m0 + wm + i * 16 + e + (hi ? 8 : 0);
                int col = n0 + wn + j * 16 + ln;
                if (OUT_BF16)
                    ((u16*)Cout)[(size_t)row * N + col] = f2bf(acc[i][j][e]);
                else
                    ((float*)Cout)[(size_t)row * N + col] = acc[i][j][e];
            }
}

// ---------------- flash attention (GQA), bf16 in / bf16 out ----------------
// grid: (S/128, HQ, B), block 256. Wave w owns q rows [q0, q0+16).
__global__ __launch_bounds__(256) void flash_attn(const u16* __restrict__ Q,
                                                  const u16* __restrict__ Kc,
                                                  const u16* __restrict__ Vc,
                                                  u16* __restrict__ O) {
    __shared__ __align__(16) u16 sK[32 * 136];   // 32 keys x 128 d, +8 pad
    __shared__ __align__(16) u16 sVt[128 * 40];  // transposed: 128 d x 32 keys, +8 pad
    __shared__ __align__(16) u16 sP[8][16 * 40]; // per-wave P scratch 16x32, +8 pad
    const int t    = threadIdx.x;
    const int lane = t & 31;
    const int wave = t >> 5;
    const int ln   = lane & 15;
    const bool hi  = lane >= 16;
    const int b  = blockIdx.z;
    const int h  = blockIdx.y;
    const int hk = h >> 2; // N_REP = 4
    const int q0 = blockIdx.x * 128 + wave * 16;

    // Q A-fragments: 16 rows x 128 d, split into 4 K=32 chunks
    v16bf qf[4];
    const u16* qrow = Q + (((size_t)(b * S_ + q0 + ln) * HQ_) + h) * HD_;
#pragma unroll
    for (int c = 0; c < 4; ++c) {
        int k0 = c * 32 + (hi ? 8 : 0);
        qf[c] = ldfrag(qrow + k0, qrow + k0 + 16);
    }

    v8f acc[8];
#pragma unroll
    for (int dt = 0; dt < 8; ++dt)
#pragma unroll
        for (int e = 0; e < 8; ++e) acc[dt][e] = 0.f;
    float rowmax[8], rowsum[8];
#pragma unroll
    for (int r = 0; r < 8; ++r) { rowmax[r] = -3.0e38f; rowsum[r] = 0.f; }

    const float scale = 0.08838834764831845f; // 1/sqrt(128)

    for (int kb = 0; kb < S_; kb += 32) {
        // K tile: async copy row-major into LDS
#pragma unroll
        for (int p = 0; p < 2; ++p) {
            int e = (p * 256 + t) * 8;
            int r = e >> 7, c = e & 127;
            size_t gsrc = (((size_t)(b * S_ + kb + r) * HKV_) + hk) * HD_ + c;
            copy16_g2l(&Kc[gsrc], &sK[r * 136 + c]);
            // V tile: transpose through VGPRs
            union { uint4 u; u16 s[8]; } vv;
            vv.u = *(const uint4*)(&Vc[gsrc]);
#pragma unroll
            for (int i = 0; i < 8; ++i) sVt[(c + i) * 40 + r] = vv.s[i];
        }
        wait_async_copies();
        __syncthreads();

        // scores: two 16x16 tiles (keys kb..kb+15, kb+16..kb+31)
        v8f s0, s1;
#pragma unroll
        for (int e = 0; e < 8; ++e) { s0[e] = 0.f; s1[e] = 0.f; }
#pragma unroll
        for (int c = 0; c < 4; ++c) {
            int koff = c * 32 + (hi ? 8 : 0);
            const u16* kp0 = &sK[ln * 136 + koff];
            s0 = wmma_bf16(qf[c], ldfrag(kp0, kp0 + 16), s0);
            const u16* kp1 = &sK[(ln + 16) * 136 + koff];
            s1 = wmma_bf16(qf[c], ldfrag(kp1, kp1 + 16), s1);
        }

        // online softmax; row r maps to (vgpr e=r, lane half) in score & out layouts
        float corr[8];
#pragma unroll
        for (int r = 0; r < 8; ++r) {
            float a = s0[r] * scale, bb = s1[r] * scale;
            float tm = fmaxf(a, bb);
#pragma unroll
            for (int m = 8; m >= 1; m >>= 1) tm = fmaxf(tm, __shfl_xor(tm, m, 16));
            float mn = fmaxf(rowmax[r], tm);
            float cr = __expf(rowmax[r] - mn);
            rowmax[r] = mn;
            float p0 = __expf(a - mn), p1 = __expf(bb - mn);
            s0[r] = p0; s1[r] = p1;
            float ts = p0 + p1;
#pragma unroll
            for (int m = 8; m >= 1; m >>= 1) ts += __shfl_xor(ts, m, 16);
            rowsum[r] = rowsum[r] * cr + ts;
            corr[r] = cr;
        }
#pragma unroll
        for (int dt = 0; dt < 8; ++dt)
#pragma unroll
            for (int r = 0; r < 8; ++r) acc[dt][r] *= corr[r];

        // transpose P (D-layout) -> A-fragment layout via wave-private LDS
        u16* my = &sP[wave][0];
#pragma unroll
        for (int r = 0; r < 8; ++r) {
            int row = r + (hi ? 8 : 0);
            my[row * 40 + ln]      = f2bf(s0[r]);
            my[row * 40 + ln + 16] = f2bf(s1[r]);
        }
        asm volatile("s_wait_dscnt 0" ::: "memory");
        const u16* pbase = my + ln * 40;
        int pk0 = hi ? 8 : 0;
        v16bf pf = ldfrag(pbase + pk0, pbase + pk0 + 16);

        // out += P(16x32) * V(32x128): 8 N-tiles of 16
#pragma unroll
        for (int dt = 0; dt < 8; ++dt) {
            const u16* vp = &sVt[(dt * 16 + ln) * 40 + (hi ? 8 : 0)];
            acc[dt] = wmma_bf16(pf, ldfrag(vp, vp + 16), acc[dt]);
        }
        __syncthreads();
    }

    // epilogue: divide by row sums, store bf16 [b,s,h,d]
#pragma unroll
    for (int dt = 0; dt < 8; ++dt)
#pragma unroll
        for (int r = 0; r < 8; ++r) {
            int row = q0 + r + (hi ? 8 : 0);
            int col = dt * 16 + ln;
            float v = acc[dt][r] / rowsum[r];
            O[(((size_t)(b * S_ + row) * HQ_) + h) * HD_ + col] = f2bf(v);
        }
}

// ---------------- host-side launcher ----------------
extern "C" void kernel_launch(void* const* d_in, const int* in_sizes, int n_in,
                              void* d_out, int out_size, void* d_ws, size_t ws_size,
                              hipStream_t stream) {
    (void)in_sizes; (void)n_in; (void)out_size; (void)ws_size;
    const float* x  = (const float*)d_in[0];
    const float* wq = (const float*)d_in[1];
    const float* wk = (const float*)d_in[2];
    const float* wv = (const float*)d_in[3];
    const float* wo = (const float*)d_in[4];
    const float* fc = (const float*)d_in[5];
    const float* fs = (const float*)d_in[6];
    // cache_k, cache_v, start_pos: start_pos==0 prefill, caches are inputs (not mutated)

    const size_t BS   = (size_t)B_ * S_;        // 4096
    const size_t nX   = BS * D_;                // 16.7M
    const size_t nWq  = (size_t)HQ_ * HD_ * D_; // 16.7M
    const size_t nWkv = (size_t)HKV_ * HD_ * D_;// 4.19M
    const size_t nQ   = BS * HQ_ * HD_;         // 16.7M
    const size_t nKV  = BS * HKV_ * HD_;        // 4.19M

    size_t off = 0;
    auto carve = [&](size_t bytes) -> char* {
        char* p = (char*)d_ws + off;
        off += (bytes + 255) & ~(size_t)255;
        return p;
    };
    u16* x_bf    = (u16*)carve(nX * 2);
    u16* wq_bf   = (u16*)carve(nWq * 2);
    u16* wk_bf   = (u16*)carve(nWkv * 2);
    u16* wv_bf   = (u16*)carve(nWkv * 2);
    u16* wo_bf   = (u16*)carve(nWq * 2);
    u16* q_bf    = (u16*)carve(nQ * 2);
    u16* k_bf    = (u16*)carve(nKV * 2);
    u16* v_bf    = (u16*)carve(nKV * 2);
    u16* attn_bf = (u16*)carve(nQ * 2);

    const int TPB = 256;
    // 1) convert to bf16
    cvt_f32_bf16<<<(int)((nX + TPB - 1) / TPB), TPB, 0, stream>>>(x, x_bf, (int)nX);
    cvt_f32_bf16<<<(int)((nWq + TPB - 1) / TPB), TPB, 0, stream>>>(wq, wq_bf, (int)nWq);
    cvt_f32_bf16<<<(int)((nWkv + TPB - 1) / TPB), TPB, 0, stream>>>(wk, wk_bf, (int)nWkv);
    cvt_f32_bf16<<<(int)((nWkv + TPB - 1) / TPB), TPB, 0, stream>>>(wv, wv_bf, (int)nWkv);
    cvt_f32_bf16<<<(int)((nWq + TPB - 1) / TPB), TPB, 0, stream>>>(wo, wo_bf, (int)nWq);

    // 2) QKV projections (bf16 out): C = A * W^T
    {
        dim3 g(HQ_ * HD_ / 128, (unsigned)(BS / 128));
        gemm_absT<true><<<g, TPB, 0, stream>>>(x_bf, wq_bf, q_bf, (int)BS, HQ_ * HD_, D_);
    }
    {
        dim3 g(HKV_ * HD_ / 128, (unsigned)(BS / 128));
        gemm_absT<true><<<g, TPB, 0, stream>>>(x_bf, wk_bf, k_bf, (int)BS, HKV_ * HD_, D_);
        gemm_absT<true><<<g, TPB, 0, stream>>>(x_bf, wv_bf, v_bf, (int)BS, HKV_ * HD_, D_);
    }

    // 3) RoPE in place on q and k
    {
        int totq = (int)(BS * HQ_ * (HD_ / 2));
        rope_bf16<<<(totq + TPB - 1) / TPB, TPB, 0, stream>>>(q_bf, fc, fs, HQ_, totq);
        int totk = (int)(BS * HKV_ * (HD_ / 2));
        rope_bf16<<<(totk + TPB - 1) / TPB, TPB, 0, stream>>>(k_bf, fc, fs, HKV_, totk);
    }

    // 4) flash attention
    {
        dim3 g(S_ / 128, HQ_, B_);
        flash_attn<<<g, TPB, 0, stream>>>(q_bf, k_bf, v_bf, attn_bf);
    }

    // 5) output projection (f32 out to d_out): out = attn * wo^T
    {
        dim3 g(D_ / 128, (unsigned)(BS / 128));
        gemm_absT<false><<<g, TPB, 0, stream>>>(attn_bf, wo_bf, (float*)d_out, (int)BS, D_, HQ_ * HD_);
    }
}
